// GraphEncoder_28887950033188
// MI455X (gfx1250) — compile-verified, matrix operand order checked by reading, not run on hardware
//
#include <hip/hip_runtime.h>

#define N_NODES 3072
#define DIM     256
#define HEADS   8
#define DHEAD   32
#define LAYERS  4
#define FFDIM   512
#define QKVW    (3 * DIM)   // 768
#define LN_EPS  1e-5f

typedef __bf16 bf16_t;
typedef __attribute__((ext_vector_type(16))) __bf16 v16bf;
typedef __attribute__((ext_vector_type(8)))  float  v8f;
typedef __attribute__((ext_vector_type(4)))  unsigned int u32x4;

struct FragBits { u32x4 lo, hi; };

__device__ __forceinline__ v8f zero_v8f() {
  v8f z = {0.f, 0.f, 0.f, 0.f, 0.f, 0.f, 0.f, 0.f};
  return z;
}

// Load one 16x32 bf16 fragment (A-layout; also serves as B since our B operands
// are stored K-contiguous per column). ISA 7.12.2: lane l -> row l&15, K-half l>>4,
// chunks K = [8*kh .. 8*kh+7] and [16+8*kh .. 16+8*kh+7]. ld must be a multiple of 8.
__device__ __forceinline__ v16bf load_frag(const bf16_t* __restrict__ base, int ld) {
  const int lane = threadIdx.x & 31;
  const bf16_t* p = base + (size_t)(lane & 15) * (size_t)ld + (size_t)((lane >> 4) * 8);
  FragBits fb;
  fb.lo = *reinterpret_cast<const u32x4*>(p);        // 16 bytes: K chunk 0
  fb.hi = *reinterpret_cast<const u32x4*>(p + 16);   // 16 bytes: K chunk 1
  return __builtin_bit_cast(v16bf, fb);
}

__device__ __forceinline__ void prefetch_frag(const bf16_t* base, int ld) {
  const int lane = threadIdx.x & 31;
  const bf16_t* p = base + (size_t)(lane & 15) * (size_t)ld + (size_t)((lane >> 4) * 8);
  __builtin_prefetch(p, 0, 3);   // -> global_prefetch_b8 (whole cacheline)
}

__device__ __forceinline__ v8f wmma_bf16(v16bf a, v16bf b, v8f c) {
  // D = A(16x32 bf16) * B(32x16 bf16) + C(16x16 f32)
  return __builtin_amdgcn_wmma_f32_16x16x32_bf16(false, a, false, b, (short)0, c,
                                                 false, false);
}

// ---------------------------------------------------------------------------
// Generic tiled bf16 GEMM: out = epilogue( A[MxK] @ Bt[NoutxK]^T )
// One wave computes a 32x64 tile (2 A frags x 4 B frags = 8 WMMAs per K step).
// K loop is unrolled by 2 with ping-pong fragment buffers: set B loads while
// set A's WMMAs execute and vice versa -> no register-copy hand-off, partial
// loadcnt waits, plus distance-1.5 global_prefetch_b8.
// REQUIRES: K % 64 == 0 (all call sites: 256, 512, 3072).
// ---------------------------------------------------------------------------
__global__ __launch_bounds__(32) void gemm_bf16_kernel(
    const bf16_t* __restrict__ A, int lda,
    const bf16_t* __restrict__ Bt, int ldb,
    const float* __restrict__ bias,      // len Nout, nullable
    const float* __restrict__ resid,     // MxNout f32, nullable (added post-ReLU)
    const float* __restrict__ rowScale,  // len M, nullable (applied first)
    float* __restrict__ outF,            // MxNout f32, nullable
    bf16_t* __restrict__ outB,           // MxNout bf16, nullable
    int M, int Nout, int K, int doRelu)
{
  const int lane = threadIdx.x & 31;
  const int i0 = blockIdx.x * 32;
  const int n0 = blockIdx.y * 64;

  const bf16_t* A0 = A + (size_t)i0 * (size_t)lda;
  const bf16_t* A1 = A + (size_t)(i0 + 16) * (size_t)lda;
  const bf16_t* B0 = Bt + (size_t)(n0 +  0) * (size_t)ldb;
  const bf16_t* B1 = Bt + (size_t)(n0 + 16) * (size_t)ldb;
  const bf16_t* B2 = Bt + (size_t)(n0 + 32) * (size_t)ldb;
  const bf16_t* B3 = Bt + (size_t)(n0 + 48) * (size_t)ldb;
  const bf16_t* Brow[4] = {B0, B1, B2, B3};

  v8f acc[2][4];
#pragma unroll
  for (int r = 0; r < 2; ++r)
#pragma unroll
    for (int t = 0; t < 4; ++t) acc[r][t] = zero_v8f();

  // Preload ping set (K step 0)
  v16bf a0A = load_frag(A0, lda);
  v16bf a1A = load_frag(A1, lda);
  v16bf bA[4];
#pragma unroll
  for (int t = 0; t < 4; ++t) bA[t] = load_frag(Brow[t], ldb);

  for (int kc = 0; kc < K; kc += 64) {
    // Load pong set (kc+32) -- always in range since K % 64 == 0.
    v16bf a0B = load_frag(A0 + kc + 32, lda);
    v16bf a1B = load_frag(A1 + kc + 32, lda);
    v16bf bB[4];
#pragma unroll
    for (int t = 0; t < 4; ++t) bB[t] = load_frag(Brow[t] + kc + 32, ldb);
    prefetch_frag(A0 + kc + 96, lda);
    prefetch_frag(A1 + kc + 96, lda);
#pragma unroll
    for (int t = 0; t < 4; ++t) prefetch_frag(Brow[t] + kc + 96, ldb);

    // WMMAs on ping set.
#pragma unroll
    for (int t = 0; t < 4; ++t) {
      acc[0][t] = wmma_bf16(a0A, bA[t], acc[0][t]);
      acc[1][t] = wmma_bf16(a1A, bA[t], acc[1][t]);
    }

    // Reload ping set (kc+64) while pong WMMAs run (uniform tail guard).
    if (kc + 64 < K) {
      a0A = load_frag(A0 + kc + 64, lda);
      a1A = load_frag(A1 + kc + 64, lda);
#pragma unroll
      for (int t = 0; t < 4; ++t) bA[t] = load_frag(Brow[t] + kc + 64, ldb);
      prefetch_frag(A0 + kc + 128, lda);
      prefetch_frag(A1 + kc + 128, lda);
#pragma unroll
      for (int t = 0; t < 4; ++t) prefetch_frag(Brow[t] + kc + 128, ldb);
    }

    // WMMAs on pong set.
#pragma unroll
    for (int t = 0; t < 4; ++t) {
      acc[0][t] = wmma_bf16(a0B, bB[t], acc[0][t]);
      acc[1][t] = wmma_bf16(a1B, bB[t], acc[1][t]);
    }
  }

  const int mb = (lane >> 4) * 8;
  const int nl = lane & 15;
#pragma unroll
  for (int r = 0; r < 2; ++r) {
#pragma unroll
    for (int t = 0; t < 4; ++t) {
#pragma unroll
      for (int v = 0; v < 8; ++v) {
        const int m = i0 + 16 * r + mb + v;
        const int n = n0 + 16 * t + nl;
        float val = acc[r][t][v];
        if (rowScale) val *= rowScale[m];
        if (bias) val += bias[n];
        if (doRelu) val = fmaxf(val, 0.0f);
        if (resid) val += resid[(size_t)m * Nout + n];
        if (outF) outF[(size_t)m * Nout + n] = val;
        if (outB) outB[(size_t)m * Nout + n] = (bf16_t)val;
      }
    }
  }
}

// ---------------------------------------------------------------------------
// Flash-style masked attention, one wave per (16-row query tile, head).
// scores masked by (adj + I) > 0; online softmax; never materializes NxN.
// ---------------------------------------------------------------------------
__global__ __launch_bounds__(32) void flash_attn_kernel(
    const bf16_t* __restrict__ qkv,    // N x 768 (q | k | v)
    const bf16_t* __restrict__ vT,     // 256 x N (v transposed)
    const bf16_t* __restrict__ adjbf,  // N x N  (adj + I)
    bf16_t* __restrict__ outB)         // N x 256 attention output
{
  __shared__ bf16_t Pt[16 * 32];

  const int lane = threadIdx.x & 31;
  const int i0 = blockIdx.x * 16;
  const int head = blockIdx.y;
  const int mb = (lane >> 4) * 8;
  const int nl = lane & 15;
  const float scale = 0.17677669529663687f;  // 1/sqrt(32)

  const v16bf qf = load_frag(qkv + (size_t)i0 * QKVW + head * DHEAD, QKVW);

  float rmax[8], rsum[8];
  v8f o0 = zero_v8f(), o1 = zero_v8f();
#pragma unroll
  for (int v = 0; v < 8; ++v) { rmax[v] = -1e30f; rsum[v] = 0.0f; }

  for (int jc = 0; jc < N_NODES; jc += 32) {
    const v16bf k0 = load_frag(qkv + (size_t)jc * QKVW + DIM + head * DHEAD, QKVW);
    const v16bf k1 = load_frag(qkv + (size_t)(jc + 16) * QKVW + DIM + head * DHEAD, QKVW);
    v8f s0 = wmma_bf16(qf, k0, zero_v8f());
    v8f s1 = wmma_bf16(qf, k1, zero_v8f());

    float a0[8], a1[8], nm[8], corr[8], p0[8], p1[8];
#pragma unroll
    for (int v = 0; v < 8; ++v) {
      const size_t row = (size_t)(i0 + mb + v) * N_NODES;
      const float m0 = (float)adjbf[row + jc + nl];
      const float m1 = (float)adjbf[row + jc + 16 + nl];
      a0[v] = (m0 > 0.0f) ? s0[v] * scale : -1e30f;
      a1[v] = (m1 > 0.0f) ? s1[v] * scale : -1e30f;
    }
#pragma unroll
    for (int v = 0; v < 8; ++v) {
      float cm = fmaxf(a0[v], a1[v]);
      cm = fmaxf(cm, __shfl_xor(cm, 1));
      cm = fmaxf(cm, __shfl_xor(cm, 2));
      cm = fmaxf(cm, __shfl_xor(cm, 4));
      cm = fmaxf(cm, __shfl_xor(cm, 8));
      nm[v] = fmaxf(rmax[v], cm);
      corr[v] = __expf(rmax[v] - nm[v]);
      rmax[v] = nm[v];
    }
#pragma unroll
    for (int v = 0; v < 8; ++v) {
      p0[v] = __expf(a0[v] - nm[v]);
      p1[v] = __expf(a1[v] - nm[v]);
      float rs = p0[v] + p1[v];
      rs += __shfl_xor(rs, 1);
      rs += __shfl_xor(rs, 2);
      rs += __shfl_xor(rs, 4);
      rs += __shfl_xor(rs, 8);
      rsum[v] = rsum[v] * corr[v] + rs;
      o0[v] *= corr[v];
      o1[v] *= corr[v];
    }
    // C-layout -> A-layout conversion through LDS (single wave, LDS is in-order)
#pragma unroll
    for (int v = 0; v < 8; ++v) {
      Pt[(mb + v) * 32 + nl] = (bf16_t)p0[v];
      Pt[(mb + v) * 32 + 16 + nl] = (bf16_t)p1[v];
    }
    const v16bf pf = load_frag(Pt, 32);
    const v16bf vf0 = load_frag(vT + (size_t)(head * DHEAD) * N_NODES + jc, N_NODES);
    const v16bf vf1 = load_frag(vT + (size_t)(head * DHEAD + 16) * N_NODES + jc, N_NODES);
    o0 = wmma_bf16(pf, vf0, o0);
    o1 = wmma_bf16(pf, vf1, o1);
  }

#pragma unroll
  for (int v = 0; v < 8; ++v) {
    const size_t row = (size_t)(i0 + mb + v) * DIM + head * DHEAD;
    outB[row + nl] = (bf16_t)(o0[v] / rsum[v]);
    outB[row + 16 + nl] = (bf16_t)(o1[v] / rsum[v]);
  }
}

// ---------------------------------------------------------------------------
// f32 LayerNorm over D=256, one block per row.
// ---------------------------------------------------------------------------
__global__ __launch_bounds__(DIM) void layernorm_kernel(
    const float* __restrict__ x, const float* __restrict__ g,
    const float* __restrict__ b, float* __restrict__ outF,
    bf16_t* __restrict__ outB)
{
  __shared__ float sbuf[DIM];
  const int row = blockIdx.x, t = threadIdx.x;
  const float val = x[(size_t)row * DIM + t];
  sbuf[t] = val; __syncthreads();
  for (int s = DIM / 2; s > 0; s >>= 1) { if (t < s) sbuf[t] += sbuf[t + s]; __syncthreads(); }
  const float mean = sbuf[0] * (1.0f / DIM);
  __syncthreads();
  const float d = val - mean;
  sbuf[t] = d * d; __syncthreads();
  for (int s = DIM / 2; s > 0; s >>= 1) { if (t < s) sbuf[t] += sbuf[t + s]; __syncthreads(); }
  const float var = sbuf[0] * (1.0f / DIM);
  const float y = d * rsqrtf(var + LN_EPS) * g[t] + b[t];
  outF[(size_t)row * DIM + t] = y;
  outB[(size_t)row * DIM + t] = (bf16_t)y;
}

// ---------------------------------------------------------------------------
// Prep kernels
// ---------------------------------------------------------------------------
__global__ __launch_bounds__(128) void deg_kernel(const float* __restrict__ adj,
                                                  float* __restrict__ deg,
                                                  float* __restrict__ dis) {
  const int lane = threadIdx.x & 31;
  const int row = blockIdx.x * 4 + (threadIdx.x >> 5);
  float s = 0.0f;
  for (int j = lane; j < N_NODES; j += 32) s += adj[(size_t)row * N_NODES + j];
  s += __shfl_xor(s, 16); s += __shfl_xor(s, 8); s += __shfl_xor(s, 4);
  s += __shfl_xor(s, 2);  s += __shfl_xor(s, 1);
  if (lane == 0) { deg[row] = s; dis[row] = rsqrtf(s + 1.0f); }
}

__global__ __launch_bounds__(256) void maxdeg_kernel(const float* __restrict__ deg,
                                                     float* __restrict__ maxd) {
  __shared__ float sbuf[256];
  const int t = threadIdx.x;
  float m = 0.0f;
  for (int i = t; i < N_NODES; i += 256) m = fmaxf(m, deg[i]);
  sbuf[t] = m; __syncthreads();
  for (int s = 128; s > 0; s >>= 1) { if (t < s) sbuf[t] = fmaxf(sbuf[t], sbuf[t + s]); __syncthreads(); }
  if (t == 0) maxd[0] = fmaxf(sbuf[0], 1.0f);
}

__global__ __launch_bounds__(256) void adjbf_kernel(const float* __restrict__ adj,
                                                    bf16_t* __restrict__ adjbf) {
  const size_t idx = (size_t)blockIdx.x * 256 + threadIdx.x;
  if (idx >= (size_t)N_NODES * N_NODES) return;
  const int r = (int)(idx / N_NODES);
  const int c = (int)(idx - (size_t)r * N_NODES);
  adjbf[idx] = (bf16_t)(adj[idx] + (r == c ? 1.0f : 0.0f));
}

// qfeatT[d][j] = dis[j] * (Wp[d,0]*1 + Wp[d,1]*deg_norm[j] + bp[d])  (bf16, 256 x N)
__global__ __launch_bounds__(256) void feat_kernel(const float* __restrict__ Wp,
                                                   const float* __restrict__ bp,
                                                   const float* __restrict__ deg,
                                                   const float* __restrict__ dis,
                                                   const float* __restrict__ maxd,
                                                   bf16_t* __restrict__ qT) {
  const int idx = blockIdx.x * 256 + threadIdx.x;
  if (idx >= DIM * N_NODES) return;
  const int d = idx / N_NODES;
  const int j = idx - d * N_NODES;
  const float dn = deg[j] / maxd[0];
  const float p = Wp[d * 2] + Wp[d * 2 + 1] * dn + bp[d];
  qT[idx] = (bf16_t)(dis[j] * p);
}

__global__ __launch_bounds__(256) void cvt_bf16_kernel(const float* __restrict__ in,
                                                       bf16_t* __restrict__ out, int count) {
  const int idx = blockIdx.x * 256 + threadIdx.x;
  if (idx < count) out[idx] = (bf16_t)in[idx];
}

// vT[c][i] = qkv[i][512 + c]
__global__ __launch_bounds__(256) void vtrans_kernel(const bf16_t* __restrict__ qkv,
                                                     bf16_t* __restrict__ vT) {
  const int idx = blockIdx.x * 256 + threadIdx.x;
  if (idx >= DIM * N_NODES) return;
  const int c = idx / N_NODES;
  const int i = idx - c * N_NODES;
  vT[idx] = qkv[(size_t)i * QKVW + 2 * DIM + c];
}

// ---------------------------------------------------------------------------
extern "C" void kernel_launch(void* const* d_in, const int* in_sizes, int n_in,
                              void* d_out, int out_size, void* d_ws, size_t ws_size,
                              hipStream_t stream) {
  const float* adj  = (const float*)d_in[0];
  const float* Wp   = (const float*)d_in[1];
  const float* bp   = (const float*)d_in[2];
  const float* Wqkv = (const float*)d_in[3];
  const float* bqkv = (const float*)d_in[4];
  const float* Wo   = (const float*)d_in[5];
  const float* bo   = (const float*)d_in[6];
  const float* g1   = (const float*)d_in[7];
  const float* b1   = (const float*)d_in[8];
  const float* W1   = (const float*)d_in[9];
  const float* c1   = (const float*)d_in[10];
  const float* W2   = (const float*)d_in[11];
  const float* c2   = (const float*)d_in[12];
  const float* g2   = (const float*)d_in[13];
  const float* b2   = (const float*)d_in[14];

  char* ws = (char*)d_ws;
  size_t off = 0;
  auto alloc = [&](size_t bytes) -> void* {
    void* p = ws + off;
    off = (off + bytes + 255) & ~(size_t)255;
    return p;
  };

  const size_t N = N_NODES;
  float*  deg    = (float*) alloc(N * 4);
  float*  dis    = (float*) alloc(N * 4);
  float*  maxd   = (float*) alloc(256);
  bf16_t* adjbf  = (bf16_t*)alloc(N * N * 2);
  bf16_t* qfeatT = (bf16_t*)alloc((size_t)DIM * N * 2);
  float*  h      = (float*) alloc(N * DIM * 4);
  bf16_t* hbf    = (bf16_t*)alloc(N * DIM * 2);
  bf16_t* qkvbf  = (bf16_t*)alloc(N * QKVW * 2);
  bf16_t* vT     = (bf16_t*)alloc((size_t)DIM * N * 2);
  bf16_t* obf    = (bf16_t*)alloc(N * DIM * 2);
  float*  tmp    = (float*) alloc(N * DIM * 4);
  bf16_t* f1bf   = (bf16_t*)alloc(N * FFDIM * 2);
  bf16_t* WqkvB  = (bf16_t*)alloc((size_t)LAYERS * QKVW * DIM * 2);
  bf16_t* WoB    = (bf16_t*)alloc((size_t)LAYERS * DIM * DIM * 2);
  bf16_t* W1B    = (bf16_t*)alloc((size_t)LAYERS * FFDIM * DIM * 2);
  bf16_t* W2B    = (bf16_t*)alloc((size_t)LAYERS * DIM * FFDIM * 2);

  // --- prep ---
  deg_kernel<<<N_NODES / 4, 128, 0, stream>>>(adj, deg, dis);
  maxdeg_kernel<<<1, 256, 0, stream>>>(deg, maxd);
  adjbf_kernel<<<(int)((N * N + 255) / 256), 256, 0, stream>>>(adj, adjbf);
  feat_kernel<<<(DIM * N_NODES + 255) / 256, 256, 0, stream>>>(Wp, bp, deg, dis, maxd, qfeatT);

  const int nWq = LAYERS * QKVW * DIM, nWo = LAYERS * DIM * DIM;
  const int nW1 = LAYERS * FFDIM * DIM, nW2 = LAYERS * DIM * FFDIM;
  cvt_bf16_kernel<<<(nWq + 255) / 256, 256, 0, stream>>>(Wqkv, WqkvB, nWq);
  cvt_bf16_kernel<<<(nWo + 255) / 256, 256, 0, stream>>>(Wo, WoB, nWo);
  cvt_bf16_kernel<<<(nW1 + 255) / 256, 256, 0, stream>>>(W1, W1B, nW1);
  cvt_bf16_kernel<<<(nW2 + 255) / 256, 256, 0, stream>>>(W2, W2B, nW2);

  // h0 = relu(dis[i] * sum_j (adj+I)[i,j] * qfeatT[:,j])
  gemm_bf16_kernel<<<dim3(N_NODES / 32, DIM / 64), 32, 0, stream>>>(
      adjbf, N_NODES, qfeatT, N_NODES, nullptr, nullptr, dis, h, hbf,
      N_NODES, DIM, N_NODES, 1);

  for (int l = 0; l < LAYERS; ++l) {
    // qkv = h @ Wqkv^T + bqkv
    gemm_bf16_kernel<<<dim3(N_NODES / 32, QKVW / 64), 32, 0, stream>>>(
        hbf, DIM, WqkvB + (size_t)l * QKVW * DIM, DIM, bqkv + (size_t)l * QKVW,
        nullptr, nullptr, nullptr, qkvbf, N_NODES, QKVW, DIM, 0);
    vtrans_kernel<<<(DIM * N_NODES + 255) / 256, 256, 0, stream>>>(qkvbf, vT);

    flash_attn_kernel<<<dim3(N_NODES / 16, HEADS), 32, 0, stream>>>(qkvbf, vT, adjbf, obf);

    // tmp = h + (attn_out @ Wo^T + bo)
    gemm_bf16_kernel<<<dim3(N_NODES / 32, DIM / 64), 32, 0, stream>>>(
        obf, DIM, WoB + (size_t)l * DIM * DIM, DIM, bo + (size_t)l * DIM,
        h, nullptr, tmp, nullptr, N_NODES, DIM, DIM, 0);
    layernorm_kernel<<<N_NODES, DIM, 0, stream>>>(
        tmp, g1 + (size_t)l * DIM, b1 + (size_t)l * DIM, h, hbf);

    // f1 = relu(h @ W1^T + c1)
    gemm_bf16_kernel<<<dim3(N_NODES / 32, FFDIM / 64), 32, 0, stream>>>(
        hbf, DIM, W1B + (size_t)l * FFDIM * DIM, DIM, c1 + (size_t)l * FFDIM,
        nullptr, nullptr, nullptr, f1bf, N_NODES, FFDIM, DIM, 1);
    // tmp = h + (f1 @ W2^T + c2)
    gemm_bf16_kernel<<<dim3(N_NODES / 32, DIM / 64), 32, 0, stream>>>(
        f1bf, FFDIM, W2B + (size_t)l * DIM * FFDIM, FFDIM, c2 + (size_t)l * DIM,
        h, nullptr, tmp, nullptr, N_NODES, DIM, FFDIM, 0);

    float* lnOut = (l == LAYERS - 1) ? (float*)d_out : h;
    layernorm_kernel<<<N_NODES, DIM, 0, stream>>>(
        tmp, g2 + (size_t)l * DIM, b2 + (size_t)l * DIM, lnOut, hbf);
  }
}
